// Attention_62663572848743
// MI455X (gfx1250) — matrix operand order repaired and checked by reading, not compiled, not run on hardware
//
#include <hip/hip_runtime.h>

// ---------------------------------------------------------------------------
// Multi-head attention forward, CDNA5 (gfx1250), wave32 + WMMA.
// B=128, N=512, C=256, H=8, hd=32.
// All four GEMMs (qkv, q@k^T, p@v, proj) run on v_wmma_f32_16x16x32_f16.
// Intermediates (q,k,v, attn-out) are f16 in workspace to halve HBM traffic;
// V is stored transposed [B][H][hd][N] so the P@V B-fragments are contiguous.
// ---------------------------------------------------------------------------

typedef __attribute__((ext_vector_type(16))) _Float16 v16h;
typedef __attribute__((ext_vector_type(8)))  _Float16 v8h;
typedef __attribute__((ext_vector_type(8)))  float    v8f;
typedef __attribute__((ext_vector_type(4)))  float    v4f;

#define WMMA_F16(a, b, c) \
    __builtin_amdgcn_wmma_f32_16x16x32_f16(false, (a), false, (b), (short)0, (c), false, false)

static constexpr int kB  = 128;
static constexpr int kN  = 512;
static constexpr int kC  = 256;
static constexpr int kH  = 8;
static constexpr int kHD = 32;
static constexpr float kScale = 0.17677669529663687f;  // 32^-0.5

// ---------------------------------------------------------------------------
// Kernel 1: QKV projection.  Y[65536, 768] = X[65536,256] @ W^T + b.
// One wave per 16x16 output tile, K-loop of 8 WMMAs.
// Writes f16:  q,k in [B][H][N][hd] (q pre-scaled by hd^-0.5),
//              v transposed in [B][H][hd][N]  (for contiguous P@V fragments).
// ISA fragment layouts (CDNA5 ISA):
//   A (16x32 f16): lane L -> row L%16; halves e=0..7 -> K = (L<16?0:8)+e,
//                  e=8..15 -> K = 16+(L<16?0:8)+(e-8).
//   B (32x16 f16): lane L -> col L%16; code loads A and B alike with K = (L<16?0:16)+e.
//   D (16x16 f32): VGPR r -> row (L<16?0:8)+r, col L%16.
// ---------------------------------------------------------------------------
__global__ __launch_bounds__(32) void qkv_wmma(
    const float* __restrict__ x, const float* __restrict__ w,
    const float* __restrict__ bias,
    _Float16* __restrict__ q, _Float16* __restrict__ k, _Float16* __restrict__ vt)
{
    const int lane = threadIdx.x;
    const int lm = lane & 15;
    const int hi = lane >> 4;
    const int mtile = blockIdx.x;                 // 65536/16 = 4096
    const int ntile = blockIdx.y;                 // 768/16   = 48

    const float* xrow = x + (size_t)(mtile * 16 + lm) * kC;   // A: row = lm
    const int    ncol = ntile * 16 + lm;                      // B/D: col = lm
    const float* wrow = w + (size_t)ncol * kC;

    v8f acc = {};
    for (int kb = 0; kb < kC; kb += 32) {
        v16h a, b;
        // A: 16 contiguous floats of X row, 64B aligned -> global_load_b128 x4
        const v4f a0 = *(const v4f*)(xrow + kb + hi * 16);
        const v4f a1 = *(const v4f*)(xrow + kb + hi * 16 + 4);
        const v4f a2 = *(const v4f*)(xrow + kb + hi * 16 + 8);
        const v4f a3 = *(const v4f*)(xrow + kb + hi * 16 + 12);
#pragma unroll
        for (int j = 0; j < 4; ++j) {
            a[j]      = (_Float16)a0[j];
            a[4 + j]  = (_Float16)a1[j];
            a[8 + j]  = (_Float16)a2[j];
            a[12 + j] = (_Float16)a3[j];
        }
        // B: 16 contiguous floats of W row, 64B aligned
        const v4f b0 = *(const v4f*)(wrow + kb + hi * 16);
        const v4f b1 = *(const v4f*)(wrow + kb + hi * 16 + 4);
        const v4f b2 = *(const v4f*)(wrow + kb + hi * 16 + 8);
        const v4f b3 = *(const v4f*)(wrow + kb + hi * 16 + 12);
#pragma unroll
        for (int j = 0; j < 4; ++j) {
            b[j]      = (_Float16)b0[j];
            b[4 + j]  = (_Float16)b1[j];
            b[8 + j]  = (_Float16)b2[j];
            b[12 + j] = (_Float16)b3[j];
        }
        acc = WMMA_F16(a, b, acc);
    }

    const int   sel  = ncol >> 8;        // 0=q 1=k 2=v
    const int   head = (ncol >> 5) & 7;
    const int   dd   = ncol & 31;
    const float bj   = bias[ncol];
    const float mult = (sel == 0) ? kScale : 1.0f;
#pragma unroll
    for (int r = 0; r < 8; ++r) {
        const int m   = mtile * 16 + hi * 8 + r;   // global token
        const int bb  = m >> 9;
        const int tok = m & 511;
        const float val = (acc[r] + bj) * mult;
        const _Float16 hv = (_Float16)val;
        if (sel == 2) {
            // V transposed: [B][H][hd][N]
            vt[(((size_t)bb * kH + head) * kHD + dd) * kN + tok] = hv;
        } else {
            _Float16* dst = (sel == 0) ? q : k;
            dst[(((size_t)bb * kH + head) * kN + tok) * kHD + dd] = hv;
        }
    }
}

// ---------------------------------------------------------------------------
// Kernel 2: attention for one (b, h, 16-query-row block) per wave.
//   S(16x512) = Q_tile(16x32) @ K^T  : 32 WMMAs -> LDS f32
//   softmax over rows (2 lanes/row, LDS reductions) -> f16 probs in LDS
//   O(16x32)  = P(16x512) @ V(512x32): 16 k-chunks x 2 col-blocks = 32 WMMAs
// V is pre-transposed, so every B-fragment is one contiguous 32B load.
// LDS: 32KB scores + 16KB probs + 128B reduce  (well under 320KB/WGP).
// ---------------------------------------------------------------------------
__global__ __launch_bounds__(32) void attn_wmma(
    const _Float16* __restrict__ q, const _Float16* __restrict__ kk,
    const _Float16* __restrict__ vt, _Float16* __restrict__ ao)
{
    __shared__ float sc[16][512];
    __shared__ alignas(32) _Float16 pf[16][512];
    __shared__ float red[32];

    const int lane = threadIdx.x;
    const int lm = lane & 15;
    const int hi = lane >> 4;
    const int bh = blockIdx.y;        // b*8 + h  (1024)
    const int qb = blockIdx.x;        // 0..31
    const size_t base = (size_t)bh * kN * kHD;   // same footprint for all three

    // --- A fragment: Q tile 16x32 (loaded once; hd == WMMA K) ---
    v16h aq;
    {
        const _Float16* qt = q + base + (size_t)(qb * 16) * kHD;
        const v8h lo = *(const v8h*)(qt + lm * kHD + hi * 16);
        const v8h ho = *(const v8h*)(qt + lm * kHD + hi * 16 + 8);
#pragma unroll
        for (int j = 0; j < 8; ++j) { aq[j] = lo[j]; aq[8 + j] = ho[j]; }
    }

    // --- scores: S = (scaled Q) @ K^T ---
    for (int jb = 0; jb < 32; ++jb) {
        const _Float16* kt = kk + base + (size_t)(jb * 16) * kHD;
        // B(32x16): col = key-in-block = lm; K(=d) = hi*16 + t, contiguous
        const v16h bk = *(const v16h*)(kt + lm * kHD + hi * 16);
        v8f s = {};
        s = WMMA_F16(aq, bk, s);
#pragma unroll
        for (int r = 0; r < 8; ++r) sc[hi * 8 + r][jb * 16 + lm] = s[r];
    }
    __syncthreads();

    // --- softmax: lane pair (lm, lm+16) shares row lm; halves of 512 cols ---
    {
        const int row = lm;
        const int c0 = hi * 256;
        float mx = -3.0e38f;
        for (int c = 0; c < 256; ++c) mx = fmaxf(mx, sc[row][c0 + c]);
        red[lane] = mx;
        __syncthreads();
        const float m2 = fmaxf(red[row], red[row + 16]);
        __syncthreads();                       // all reads of red done
        float sum = 0.0f;
        for (int c = 0; c < 256; ++c) {
            const float e = __expf(sc[row][c0 + c] - m2);
            sc[row][c0 + c] = e;
            sum += e;
        }
        red[lane] = sum;
        __syncthreads();
        const float inv = 1.0f / (red[row] + red[row + 16]);
        for (int c = 0; c < 256; ++c)
            pf[row][c0 + c] = (_Float16)(sc[row][c0 + c] * inv);
    }
    __syncthreads();

    // --- O = P @ V  (V transposed: vt[.. d ..][key]) ---
    const _Float16* vhead = vt + base;   // [hd][N] for this (b,h)
    v8f o0 = {}, o1 = {};
    for (int kb = 0; kb < 16; ++kb) {          // 512 keys / 32 per WMMA
        v16h ap;
        const v8h lo = *(const v8h*)(&pf[lm][kb * 32 + hi * 16]);
        const v8h ho = *(const v8h*)(&pf[lm][kb * 32 + hi * 16 + 8]);
#pragma unroll
        for (int j = 0; j < 8; ++j) { ap[j] = lo[j]; ap[8 + j] = ho[j]; }

        // B(32x16): col = d; K = key = kb*32 + hi*16 + t -> contiguous in vt
        const v16h b0 = *(const v16h*)(vhead + (size_t)lm * kN        + kb * 32 + hi * 16);
        const v16h b1 = *(const v16h*)(vhead + (size_t)(lm + 16) * kN + kb * 32 + hi * 16);
        o0 = WMMA_F16(ap, b0, o0);
        o1 = WMMA_F16(ap, b1, o1);
    }

    // --- store attn-out as f16 [B*N][C], features h*32 + d ---
    const int b = bh >> 3, h = bh & 7;
#pragma unroll
    for (int r = 0; r < 8; ++r) {
        const int tok = qb * 16 + hi * 8 + r;
        const size_t m = (size_t)b * kN + tok;
        ao[m * kC + h * kHD + lm]      = (_Float16)o0[r];
        ao[m * kC + h * kHD + 16 + lm] = (_Float16)o1[r];
    }
}

// ---------------------------------------------------------------------------
// Kernel 3: output projection.  out[65536,256] = AO(f16) @ proj_w^T + proj_b.
// ---------------------------------------------------------------------------
__global__ __launch_bounds__(32) void proj_wmma(
    const _Float16* __restrict__ a, const float* __restrict__ w,
    const float* __restrict__ bias, float* __restrict__ out)
{
    const int lane = threadIdx.x;
    const int lm = lane & 15;
    const int hi = lane >> 4;
    const int mtile = blockIdx.x;              // 4096
    const int ntile = blockIdx.y;              // 16

    const _Float16* arow = a + (size_t)(mtile * 16 + lm) * kC;
    const float*    wrow = w + (size_t)(ntile * 16 + lm) * kC;

    v8f acc = {};
    for (int kb = 0; kb < kC; kb += 32) {
        v16h af, bf;
        const v8h lo = *(const v8h*)(arow + kb + hi * 16);
        const v8h ho = *(const v8h*)(arow + kb + hi * 16 + 8);
#pragma unroll
        for (int j = 0; j < 8; ++j) { af[j] = lo[j]; af[8 + j] = ho[j]; }
        const v4f b0 = *(const v4f*)(wrow + kb + hi * 16);
        const v4f b1 = *(const v4f*)(wrow + kb + hi * 16 + 4);
        const v4f b2 = *(const v4f*)(wrow + kb + hi * 16 + 8);
        const v4f b3 = *(const v4f*)(wrow + kb + hi * 16 + 12);
#pragma unroll
        for (int j = 0; j < 4; ++j) {
            bf[j]      = (_Float16)b0[j];
            bf[4 + j]  = (_Float16)b1[j];
            bf[8 + j]  = (_Float16)b2[j];
            bf[12 + j] = (_Float16)b3[j];
        }
        acc = WMMA_F16(af, bf, acc);
    }

    const float bj = bias[ntile * 16 + lm];
#pragma unroll
    for (int r = 0; r < 8; ++r) {
        const int m = mtile * 16 + hi * 8 + r;
        out[(size_t)m * kC + ntile * 16 + lm] = acc[r] + bj;
    }
}

// ---------------------------------------------------------------------------
extern "C" void kernel_launch(void* const* d_in, const int* in_sizes, int n_in,
                              void* d_out, int out_size, void* d_ws, size_t ws_size,
                              hipStream_t stream)
{
    const float* x      = (const float*)d_in[0];
    const float* qkv_w  = (const float*)d_in[1];
    const float* qkv_b  = (const float*)d_in[2];
    const float* proj_w = (const float*)d_in[3];
    const float* proj_b = (const float*)d_in[4];
    float* out = (float*)d_out;

    const size_t per = (size_t)kB * kH * kN * kHD;   // 16,777,216 halves each
    _Float16* q  = (_Float16*)d_ws;
    _Float16* k  = q + per;
    _Float16* vt = k + per;                          // transposed V
    _Float16* ao = vt + per;                         // total 128 MB f16

    // 1) QKV projection: 4096 x 48 tiles, one wave each.
    qkv_wmma<<<dim3(4096, 48), dim3(32), 0, stream>>>(x, qkv_w, qkv_b, q, k, vt);
    // 2) Attention: 32 query blocks x (B*H)=1024.
    attn_wmma<<<dim3(32, 1024), dim3(32), 0, stream>>>(q, k, vt, ao);
    // 3) Output projection: 4096 x 16 tiles.
    proj_wmma<<<dim3(4096, 16), dim3(32), 0, stream>>>(ao, proj_w, proj_b, out);
}
